// DerivNet2D_4320737099970
// MI455X (gfx1250) — matrix operand order repaired and checked by reading, not compiled, not run on hardware
//
#include <hip/hip_runtime.h>

typedef __attribute__((ext_vector_type(16))) _Float16 v16h;
typedef __attribute__((ext_vector_type(8)))  _Float16 v8h;
typedef __attribute__((ext_vector_type(8)))  float    v8f;

#define H1 1024
#define H2 512
#define MTILE 16                 // samples per workgroup -> 48 augmented rows
#define KCHUNK 256               // K staged through LDS per iteration
#define ASTRIDE (KCHUNK + 8)     // halves; 528B row stride -> 4-bank skew, conflict-free

// ---------------------------------------------------------------------------
// Pre-pass: convert W2 (f32, [512,1024] row-major) into f16, pre-swizzled into
// the B-fragment order this kernel uses (contiguous K per lane half):
//   addr = (((ct*32 + kstep)*32 + lane)*16 + j) halves
//   lane 0-15:  N = ct*16 + lane,     K = kstep*32 + j        (j = 0..15)
//   lane 16-31: N = ct*16 + lane-16,  K = kstep*32 + 16 + j
// so each wave's B load per WMMA step is 1KB fully contiguous.
// ---------------------------------------------------------------------------
__global__ void w2_swizzle_f16(const float* __restrict__ W2,
                               _Float16* __restrict__ W2h) {
  int o    = blockIdx.x * blockDim.x + threadIdx.x;  // 0 .. 32*32*32*16-1
  int j    = o & 15;
  int lane = (o >> 4) & 31;
  int ks   = (o >> 9) & 31;
  int ct   = o >> 14;
  int n = ct * 16 + (lane & 15);
  int k = (ks << 5) + ((lane >> 4) << 4) + j;
  W2h[o] = (_Float16)W2[n * H1 + k];
}

// ---------------------------------------------------------------------------
// Fused MLP + forward-mode derivative kernel.
// One block = 256 threads (8 waves) handles MTILE=16 samples.
// Augmented A = [z1 ; g1*W1col0 ; g1*W1col1]  (48 x 1024, f16 in LDS chunks)
// GEMM:  [48 x 1024] x [1024 x 512] via v_wmma_f32_16x16x32_f16.
// ---------------------------------------------------------------------------
__global__ void __launch_bounds__(256)
derivnet_fused(const float* __restrict__ x,
               const float* __restrict__ W1,
               const float* __restrict__ b1,
               const float* __restrict__ b2,
               const float* __restrict__ W3,
               const float* __restrict__ b3,
               const _Float16* __restrict__ W2h,
               float* __restrict__ out, int nx) {
  __shared__ _Float16 Ach[48 * ASTRIDE];     // ~25.3 KB
  __shared__ float part[3][32][MTILE];       // 6 KB

  const int tid  = threadIdx.x;
  const int m0   = blockIdx.x * MTILE;
  const int lane = tid & 31;
  const int wid  = tid >> 5;
  const int nloc = lane & 15;
  const int hi   = lane >> 4;                // which half of the wave

  // Persistent accumulators: 4 column-tiles x {h2, u1, u2} row-tiles.
  v8f accH[4], accU1[4], accU2[4];
  #pragma unroll
  for (int q = 0; q < 4; ++q)
    #pragma unroll
    for (int r = 0; r < 8; ++r) { accH[q][r] = 0.f; accU1[q][r] = 0.f; accU2[q][r] = 0.f; }

  for (int kc = 0; kc < H1 / KCHUNK; ++kc) {
    // ---- Phase 1 (VALU): layer-1 + tanh, build augmented A chunk in LDS ----
    for (int idx = tid; idx < MTILE * KCHUNK; idx += 256) {
      int jl = idx & (KCHUNK - 1);
      int m  = idx >> 8;                     // KCHUNK == 256
      int j  = kc * KCHUNK + jl;
      float x0  = x[(m0 + m) * 2 + 0];
      float x1  = x[(m0 + m) * 2 + 1];
      float w10 = W1[2 * j + 0];
      float w11 = W1[2 * j + 1];
      float h = fmaf(x0, w10, fmaf(x1, w11, b1[j]));
      float z = tanhf(h);
      float g = 1.0f - z * z;                // sech^2(h)
      Ach[(m)      * ASTRIDE + jl] = (_Float16)z;
      Ach[(16 + m) * ASTRIDE + jl] = (_Float16)(g * w10);
      Ach[(32 + m) * ASTRIDE + jl] = (_Float16)(g * w11);
    }
    __syncthreads();

    // ---- Phase 2 (WMMA): accumulate this K chunk ----
    #pragma unroll
    for (int q = 0; q < 4; ++q) {
      const int ct = wid + 8 * q;            // column tile owned by this wave
      const _Float16* bp =
          W2h + ((size_t)ct * 32 + (size_t)kc * 8) * (32 * 16) + (size_t)lane * 16;
      for (int k0 = 0; k0 < KCHUNK; k0 += 32) {
        // B fragment: 32 contiguous bytes per lane (pre-swizzled layout)
        const _Float16* bs = bp + (k0 >> 5) * (32 * 16);
        v8h blo = *(const v8h*)(bs);
        v8h bhi = *(const v8h*)(bs + 8);
        v16h bf;
        #pragma unroll
        for (int i = 0; i < 8; ++i) { bf[i] = blo[i]; bf[8 + i] = bhi[i]; }

        // A fragments for the 3 row tiles; lane holds
        // K = [base, base+16), base = k0 + hi*16 (matches B swizzle).
        const int abase = k0 + hi * 16;
        #pragma unroll
        for (int rt = 0; rt < 3; ++rt) {
          const _Float16* ap = &Ach[(rt * 16 + nloc) * ASTRIDE + abase];
          v8h alo = *(const v8h*)(ap);
          v8h ahi = *(const v8h*)(ap + 8);
          v16h af;
          #pragma unroll
          for (int i = 0; i < 8; ++i) { af[i] = alo[i]; af[8 + i] = ahi[i]; }
          v8f& acc = (rt == 0) ? accH[q] : (rt == 1) ? accU1[q] : accU2[q];
          acc = __builtin_amdgcn_wmma_f32_16x16x32_f16(
              /*neg_a=*/false, af, /*neg_b=*/false, bf,
              /*c_mod=*/(short)0, acc, /*reuse_a=*/false, /*reuse_b=*/false);
        }
      }
    }
    __syncthreads();                         // A chunk gets overwritten next iter
  }

  // ---- Epilogue: tanh layer 2, derivative scaling, W3 dot (reduce over N) ----
  // C/D layout: VGPR r, lane l -> M = r + (l>=16)*8, N = l&15.
  #pragma unroll
  for (int q = 0; q < 4; ++q) {
    const int ct  = wid + 8 * q;
    const int n   = ct * 16 + nloc;
    const float b2n = b2[n];
    const float w3n = W3[n];
    #pragma unroll
    for (int r = 0; r < 8; ++r) {
      float h2 = accH[q][r] + b2n;
      float z2 = tanhf(h2);
      float g2 = 1.0f - z2 * z2;
      float py = w3n * z2;                   // contribution to y
      float p1 = w3n * (g2 * accU1[q][r]);   // contribution to dydx1
      float p2 = w3n * (g2 * accU2[q][r]);   // contribution to dydx2
      #pragma unroll
      for (int off = 8; off >= 1; off >>= 1) {   // reduce over the 16 N-lanes
        py += __shfl_xor(py, off, 32);
        p1 += __shfl_xor(p1, off, 32);
        p2 += __shfl_xor(p2, off, 32);
      }
      if (nloc == 0) {
        int m = r + hi * 8;
        part[0][ct][m] = py;
        part[1][ct][m] = p1;
        part[2][ct][m] = p2;
      }
    }
  }
  __syncthreads();

  // Deterministic cross-column-tile reduction + final outputs.
  if (tid < MTILE) {
    float sy = 0.f, s1 = 0.f, s2 = 0.f;
    for (int ct = 0; ct < 32; ++ct) {
      sy += part[0][ct][tid];
      s1 += part[1][ct][tid];
      s2 += part[2][ct][tid];
    }
    int gm = m0 + tid;
    out[gm]              = sy + b3[0];   // y
    out[nx + gm]         = s2;           // v1 =  dydx2
    out[2 * nx + gm]     = -s1;          // v2 = -dydx1
  }
}

// ---------------------------------------------------------------------------
extern "C" void kernel_launch(void* const* d_in, const int* in_sizes, int n_in,
                              void* d_out, int out_size, void* d_ws, size_t ws_size,
                              hipStream_t stream) {
  (void)n_in; (void)out_size; (void)ws_size;
  const float* x  = (const float*)d_in[0];
  const float* W1 = (const float*)d_in[1];
  const float* b1 = (const float*)d_in[2];
  const float* W2 = (const float*)d_in[3];
  const float* b2 = (const float*)d_in[4];
  const float* W3 = (const float*)d_in[5];
  const float* b3 = (const float*)d_in[6];
  float* out = (float*)d_out;
  const int nx = in_sizes[0] / 2;

  _Float16* W2h = (_Float16*)d_ws;           // 512*1024*2 = 1 MB of scratch

  w2_swizzle_f16<<<(H2 * H1) / 256, 256, 0, stream>>>(W2, W2h);
  derivnet_fused<<<nx / MTILE, 256, 0, stream>>>(x, W1, b1, b2, W3, b3, W2h, out, nx);
}
